// LSTMMOE_52527450030670
// MI455X (gfx1250) — matrix operand order repaired and checked by reading, not compiled, not run on hardware
//
#include <hip/hip_runtime.h>

#define S_LEN 32768
#define D_CTX 1024
#define ND_   512
#define IN_   2048
#define G4H   128
#define KP    64            // K advanced per GEMM iteration (2 WMMA chunks)
#define NPAIR (IN_/KP)      // 32 iterations

typedef __attribute__((ext_vector_type(16))) __bf16 v16bf;
typedef __attribute__((ext_vector_type(8)))  float  v8f;

__device__ __forceinline__ float lrelu(float x){ return x > 0.f ? x : 0.01f*x; }
__device__ __forceinline__ float dot4f(float4 a, float4 b){
  return fmaf(a.x,b.x, fmaf(a.y,b.y, fmaf(a.z,b.z, a.w*b.w)));
}
__device__ __forceinline__ float4 lr4(float s, float4 w, float4 b){
  float4 r;
  r.x = lrelu(fmaf(s,w.x,b.x)); r.y = lrelu(fmaf(s,w.y,b.y));
  r.z = lrelu(fmaf(s,w.z,b.z)); r.w = lrelu(fmaf(s,w.w,b.w));
  return r;
}
__device__ __forceinline__ v16bf pack16(float4 a, float4 b, float4 c, float4 d){
  v16bf r;
  r[0]=(__bf16)a.x;  r[1]=(__bf16)a.y;  r[2]=(__bf16)a.z;  r[3]=(__bf16)a.w;
  r[4]=(__bf16)b.x;  r[5]=(__bf16)b.y;  r[6]=(__bf16)b.z;  r[7]=(__bf16)b.w;
  r[8]=(__bf16)c.x;  r[9]=(__bf16)c.y;  r[10]=(__bf16)c.z; r[11]=(__bf16)c.w;
  r[12]=(__bf16)d.x; r[13]=(__bf16)d.y; r[14]=(__bf16)d.z; r[15]=(__bf16)d.w;
  return r;
}
// 16 A-elements of one lane-half: floats at [0,8) and [16,24) from base
__device__ __forceinline__ void loadArun(const float* __restrict__ base, float4* d){
  d[0]=*(const float4*)(base+0);  d[1]=*(const float4*)(base+4);
  d[2]=*(const float4*)(base+16); d[3]=*(const float4*)(base+20);
}
__device__ __forceinline__ void loadApair(int kb, int mrow, int hiA,
    const float* __restrict__ ctxt, const float* __restrict__ wf,
    const float* __restrict__ bfv,  const float* __restrict__ we,
    const float* __restrict__ bev,  float4* araw, float4* abraw){
  #pragma unroll
  for (int c = 0; c < 2; ++c){
    const int k = kb + c*32 + hiA;
    if (kb < D_CTX){
      loadArun(ctxt + (size_t)mrow*D_CTX + k, &araw[c*4]);
    } else if (kb < D_CTX + ND_){
      loadArun(wf  + (k - D_CTX), &araw[c*4]);
      loadArun(bfv + (k - D_CTX), &abraw[c*4]);
    } else {
      loadArun(we  + (k - D_CTX - ND_), &araw[c*4]);
      loadArun(bev + (k - D_CTX - ND_), &abraw[c*4]);
    }
  }
}
__device__ __forceinline__ v16bf cvtAchunk(int kb, int c, float fs, float es,
                                           const float4* araw, const float4* abraw){
  const int c4 = c*4;
  if (kb < D_CTX){
    return pack16(araw[c4+0], araw[c4+1], araw[c4+2], araw[c4+3]);
  } else {
    const float s = (kb < D_CTX + ND_) ? fs : es;
    return pack16(lr4(s,araw[c4+0],abraw[c4+0]), lr4(s,araw[c4+1],abraw[c4+1]),
                  lr4(s,araw[c4+2],abraw[c4+2]), lr4(s,araw[c4+3],abraw[c4+3]));
  }
}
// B raw: floats at [0,8) and [16,24) from base per chunk
__device__ __forceinline__ void loadBpair(const float* __restrict__ wrow, int kb, float4* braw){
  #pragma unroll
  for (int c = 0; c < 2; ++c){
    const float* p = wrow + kb + c*32;
    braw[c*4+0]=*(const float4*)(p+0);  braw[c*4+1]=*(const float4*)(p+4);
    braw[c*4+2]=*(const float4*)(p+16); braw[c*4+3]=*(const float4*)(p+20);
  }
}

// -------- Kernel 1: pre0 = x @ w_ih0^T + b_ih0 (bf16 WMMA, f32 accumulate) -------
// block = 256 thr (8 waves); wave w: 16-row M-tile x all 128 cols.
// Double-buffered LDS B pairs; next pair's global loads issued before the WMMA chain.
__global__ __launch_bounds__(256) void pre0_gemm(
    const float* __restrict__ ctxt, const float* __restrict__ freq,
    const float* __restrict__ fert, const float* __restrict__ wf,
    const float* __restrict__ bfv,  const float* __restrict__ we,
    const float* __restrict__ bev,  const float* __restrict__ w_ih0,
    const float* __restrict__ b_ih0, float* __restrict__ pre0)
{
  __shared__ __align__(32) __bf16 Bs[2][2][8][32][16];   // [buf][chunk][nt][lane][k] 32 KB
  const int lane = threadIdx.x & 31;
  const int wv   = threadIdx.x >> 5;
  const int s0   = (blockIdx.x*8 + wv)*16;
  const int mrow = s0 + (lane & 15);
  const int hiA  = (lane & 16) ? 8 : 0;
  const float fs = freq[mrow];
  const float es = fert[mrow];

  v8f acc[8] = {};

  const int nB  = wv*16 + (lane & 15);
  const int hiB = (lane & 16) ? 8 : 0;
  const float* wrow = w_ih0 + (size_t)nB*IN_ + hiB;

  float4 braw[8], araw[8], abraw[8];

  // prologue: pair 0 staged + converted
  loadBpair(wrow, 0, braw);
  loadApair(0, mrow, hiA, ctxt, wf, bfv, we, bev, araw, abraw);
  *(v16bf*)&Bs[0][0][wv][lane][0] = pack16(braw[0],braw[1],braw[2],braw[3]);
  *(v16bf*)&Bs[0][1][wv][lane][0] = pack16(braw[4],braw[5],braw[6],braw[7]);
  v16bf a0 = cvtAchunk(0, 0, fs, es, araw, abraw);
  v16bf a1 = cvtAchunk(0, 1, fs, es, araw, abraw);

  for (int p = 0; p < NPAIR; ++p){
    __syncthreads();                         // Bs[p&1] visible; prior reads retired
    const bool more = (p + 1) < NPAIR;
    const int kbn = (p + 1)*KP;
    if (more){                               // issue next pair's global loads NOW
      loadBpair(wrow, kbn, braw);
      loadApair(kbn, mrow, hiA, ctxt, wf, bfv, we, bev, araw, abraw);
    }
    // 16 WMMAs on current pair (LDS B, register A) — hides the loads above
    #pragma unroll
    for (int nt = 0; nt < 8; ++nt){
      v16bf bv = *(const v16bf*)&Bs[p&1][0][nt][lane][0];
      acc[nt] = __builtin_amdgcn_wmma_f32_16x16x32_bf16(
          false, a0, false, bv, (short)0, acc[nt], false, false);
    }
    #pragma unroll
    for (int nt = 0; nt < 8; ++nt){
      v16bf bv = *(const v16bf*)&Bs[p&1][1][nt][lane][0];
      acc[nt] = __builtin_amdgcn_wmma_f32_16x16x32_bf16(
          false, a1, false, bv, (short)0, acc[nt], false, false);
    }
    if (more){                               // convert + stage into the other buffer
      *(v16bf*)&Bs[(p+1)&1][0][wv][lane][0] = pack16(braw[0],braw[1],braw[2],braw[3]);
      *(v16bf*)&Bs[(p+1)&1][1][wv][lane][0] = pack16(braw[4],braw[5],braw[6],braw[7]);
      a0 = cvtAchunk(kbn, 0, fs, es, araw, abraw);
      a1 = cvtAchunk(kbn, 1, fs, es, araw, abraw);
    }
  }

  // D layout: vgpr r, lanes 0-15 -> (M=r, N=lane), lanes 16-31 -> (M=r+8, N=lane-16)
  #pragma unroll
  for (int nt = 0; nt < 8; ++nt){
    int n = nt*16 + (lane & 15);
    float bias = b_ih0[n];
    #pragma unroll
    for (int r = 0; r < 8; ++r){
      int row = s0 + r + ((lane & 16) ? 8 : 0);
      pre0[(size_t)row*G4H + n] = acc[nt][r] + bias;
    }
  }
}

// -------- Kernel 2: persistent 4-stage pipelined LSTM scan + pred head ----------
// 13 waves: groups 0..2 = 4 waves (one lane per gate row), group 3 = head wave.
// group g processes timestep t = i - g; ONE __syncthreads per pipeline tick.
__global__ __launch_bounds__(416, 1) void lstm_scan(
    const float* __restrict__ pre0,
    const float* __restrict__ w_hh0, const float* __restrict__ b_hh0,
    const float* __restrict__ w_ih1, const float* __restrict__ w_hh1,
    const float* __restrict__ b_ih1, const float* __restrict__ b_hh1,
    const float* __restrict__ w_ih2, const float* __restrict__ w_hh2,
    const float* __restrict__ b_ih2, const float* __restrict__ b_hh2,
    const float* __restrict__ w_pred, const float* __restrict__ b_pred,
    float* __restrict__ out)
{
  __shared__ float hbuf[2][3][32];                 // double-buffered layer outputs
  const int tid  = threadIdx.x;
  const int lane = tid & 31;
  const int grp  = tid >> 7;                       // 0,1,2 compute; 3 = head wave
  const int wg   = (tid >> 5) & 3;                 // wave within group

  if (tid < 192) ((float*)hbuf)[tid] = 0.f;       // h init = 0 (both slots)

  // lane -> (element j, gate): lanes 4j..4j+3 of a wave hold gates i,f,g,o of j
  const int j    = wg*8 + (lane >> 2);
  const int gate = lane & 3;
  const int r    = gate*32 + j;                    // gate row in 4H (PyTorch i,f,g,o)

  float4 whv[8], wiv[8];
  float bsum = 0.f, c = 0.f, pre_cur = 0.f;
  float w0p = 0.f, w1p = 0.f, b0 = 0.f, b1 = 0.f;

  if (grp < 3){
    const float* whh = (grp==0) ? w_hh0 : (grp==1) ? w_hh1 : w_hh2;
    const float* bhh = (grp==0) ? b_hh0 : (grp==1) ? b_hh1 : b_hh2;
    #pragma unroll
    for (int q = 0; q < 8; ++q) whv[q] = *(const float4*)(whh + r*32 + q*4);
    bsum = bhh[r];
    if (grp >= 1){
      const float* wih = (grp==1) ? w_ih1 : w_ih2;
      const float* bih = (grp==1) ? b_ih1 : b_ih2;
      #pragma unroll
      for (int q = 0; q < 8; ++q) wiv[q] = *(const float4*)(wih + r*32 + q*4);
      bsum += bih[r];
    } else {
      pre_cur = pre0[r];                           // prefetch t = 0
    }
  } else {
    w0p = w_pred[lane]; w1p = w_pred[32 + lane];
    b0  = b_pred[0];    b1  = b_pred[1];
  }
  __syncthreads();

  for (int i = 0; i < S_LEN + 3; ++i){
    const int cur = i & 1, prv = (i + 1) & 1;
    if (grp < 3){
      const int t = i - grp;
      if (t >= 0 && t < S_LEN){
        float pre_next = 0.f;
        if (grp == 0 && (t + 1) < S_LEN)           // prefetch next step's pre0
          pre_next = pre0[(size_t)(t + 1)*G4H + r];

        float acc = bsum + ((grp == 0) ? pre_cur : 0.f);
        {
          const float4* hop = (const float4*)&hbuf[prv][grp][0];   // h_own (broadcast)
          #pragma unroll
          for (int q = 0; q < 8; ++q) acc += dot4f(whv[q], hop[q]);
        }
        if (grp >= 1){
          const float4* hip = (const float4*)&hbuf[prv][grp-1][0]; // lower-layer h
          #pragma unroll
          for (int q = 0; q < 8; ++q) acc += dot4f(wiv[q], hip[q]);
        }
        // one exp + one rcp covers sigmoid (gates i,f,o) and tanh (gate g):
        //   sigmoid(a) = 1/(1+e^-a) ;  tanh(a) = 1 - 2/(1+e^{2a})
        float arg = (gate == 2) ? 2.f*acc : -acc;
        float rp  = __builtin_amdgcn_rcpf(1.f + __expf(arg));
        float act = (gate == 2) ? fmaf(-2.f, rp, 1.f) : rp;

        const int b4 = lane & ~3;                  // gather this element's 4 gates
        float ai = __shfl(act, b4+0);
        float af = __shfl(act, b4+1);
        float ag = __shfl(act, b4+2);
        float ao = __shfl(act, b4+3);
        c = fmaf(af, c, ai*ag);
        // tanh(c) = 1 - 2/(1+e^{2c})
        float tc = fmaf(-2.f, __builtin_amdgcn_rcpf(1.f + __expf(2.f*c)), 1.f);
        float hn = ao * tc;
        if ((lane & 3) == 0) hbuf[cur][grp][j] = hn;
        if (grp == 0) pre_cur = pre_next;
      }
    } else {
      const int t3 = i - 3;                        // head: logits + log_softmax
      if (t3 >= 0 && t3 < S_LEN){
        float hv = hbuf[prv][2][lane];
        float p0 = hv*w0p, p1 = hv*w1p;
        #pragma unroll
        for (int m = 16; m >= 1; m >>= 1){
          p0 += __shfl_xor(p0, m);
          p1 += __shfl_xor(p1, m);
        }
        if (lane == 0){
          float l0 = p0 + b0, l1 = p1 + b1;
          float mx = fmaxf(l0, l1);
          float z  = mx + __logf(__expf(l0 - mx) + __expf(l1 - mx));
          out[(size_t)t3*2 + 0] = l0 - z;
          out[(size_t)t3*2 + 1] = l1 - z;
        }
      }
    }
    __syncthreads();
  }
}

extern "C" void kernel_launch(void* const* d_in, const int* in_sizes, int n_in,
                              void* d_out, int out_size, void* d_ws, size_t ws_size,
                              hipStream_t stream)
{
  (void)in_sizes; (void)n_in; (void)out_size; (void)ws_size;
  const float* ctxt   = (const float*)d_in[0];
  const float* freq   = (const float*)d_in[1];
  const float* fert   = (const float*)d_in[2];
  const float* wf     = (const float*)d_in[3];
  const float* bfv    = (const float*)d_in[4];
  const float* we     = (const float*)d_in[5];
  const float* bev    = (const float*)d_in[6];
  const float* w_pred = (const float*)d_in[7];
  const float* b_pred = (const float*)d_in[8];
  const float* w_ih0  = (const float*)d_in[9];
  const float* w_hh0  = (const float*)d_in[10];
  const float* b_ih0  = (const float*)d_in[11];
  const float* b_hh0  = (const float*)d_in[12];
  const float* w_ih1  = (const float*)d_in[13];
  const float* w_hh1  = (const float*)d_in[14];
  const float* b_ih1  = (const float*)d_in[15];
  const float* b_hh1  = (const float*)d_in[16];
  const float* w_ih2  = (const float*)d_in[17];
  const float* w_hh2  = (const float*)d_in[18];
  const float* b_ih2  = (const float*)d_in[19];
  const float* b_hh2  = (const float*)d_in[20];
  float* out  = (float*)d_out;
  float* pre0 = (float*)d_ws;                      // 32768*128 f32 = 16 MB scratch

  pre0_gemm<<<S_LEN/128, 256, 0, stream>>>(ctxt, freq, fert, wf, bfv, we, bev,
                                           w_ih0, b_ih0, pre0);
  lstm_scan<<<1, 416, 0, stream>>>(pre0, w_hh0, b_hh0,
                                   w_ih1, w_hh1, b_ih1, b_hh1,
                                   w_ih2, w_hh2, b_ih2, b_hh2,
                                   w_pred, b_pred, out);
}
